// Net_8607114461641
// MI455X (gfx1250) — matrix operand order repaired and checked by reading, not compiled, not run on hardware
//
#include <hip/hip_runtime.h>
#include <hip/hip_bf16.h>
#include <stdint.h>

typedef __attribute__((ext_vector_type(16))) _Float16 v16h;
typedef __attribute__((ext_vector_type(8)))  float    v8f;
typedef __attribute__((ext_vector_type(4)))  uint32_t v4u;

#define B_TOTAL   131072
#define NBLK      3
#define SSZ       16
#define NACT      9
#define SA        25      // SSZ+NACT
#define VD        128
#define BT        16      // batch elements per wave
#define ROWS      48      // BT*NBLK
#define OUTD      51

// d_ws dword layout: fragment-native f16 weight panels.
// Each fragment f = 256 dwords: lane l owns dwords [f*256 + l*8, +8) = its v16h B-operand.
#define WS_WQ     0        // 8 frags (nt)          -> 2048 dwords
#define WS_WK     2048
#define WS_WV     4096
#define WS_W1     6144     // 32 frags (nt*4+ks)    -> 8192 dwords
#define WS_W2     14336    // 4 frags (ks)          -> 1024 dwords
#define WS_TOTAL  15360

union H2 { uint32_t u; _Float16 h[2]; };

__device__ __forceinline__ uint32_t packh2(float lo, float hi) {
  H2 p; p.h[0] = (_Float16)lo; p.h[1] = (_Float16)hi; return p.u;
}

// ---------------- weight prep: f32 [N][K] -> fragment-native f16 panels ----------------
// B-fragment element mapping (16x16x32 f16): lane l, dword d -> n = 16*nt + (l&15),
// k = 16*(l>>4) + 2*d + 32*ks, packed {W[n][k], W[n][k+1]} (zero past Kreal).
__global__ void prep_weights(const float* __restrict__ Wq, const float* __restrict__ Wk,
                             const float* __restrict__ Wv, const float* __restrict__ W1,
                             const float* __restrict__ W2, uint32_t* __restrict__ wsU) {
  int idx = blockIdx.x * blockDim.x + threadIdx.x;
  if (idx >= WS_TOTAL) return;
  const float* W; int Kreal, n, k;
  if (idx < WS_W1) {                       // Wq/Wk/Wv: N=128, Kreal=25 (pad 32), ks=0
    int w = idx >> 11;
    int r = idx & 2047;
    int f = r >> 8, rem = r & 255;
    int l = rem >> 3, d = rem & 7;
    W = (w == 0) ? Wq : ((w == 1) ? Wk : Wv);
    Kreal = SA;
    n = 16 * f + (l & 15);
    k = 16 * (l >> 4) + 2 * d;
  } else if (idx < WS_W2) {                // W1: N=128, K=128, f = nt*4+ks
    int r = idx - WS_W1;
    int f = r >> 8, rem = r & 255;
    int l = rem >> 3, d = rem & 7;
    int nt = f >> 2, ks = f & 3;
    W = W1; Kreal = VD;
    n = 16 * nt + (l & 15);
    k = 16 * (l >> 4) + 2 * d + 32 * ks;
  } else {                                 // W2: N=16, K=128, f = ks
    int r = idx - WS_W2;
    int f = r >> 8, rem = r & 255;
    int l = rem >> 3, d = rem & 7;
    W = W2; Kreal = VD;
    n = (l & 15);
    k = 16 * (l >> 4) + 2 * d + 32 * f;
  }
  float lo = (k     < Kreal) ? W[n * Kreal + k]     : 0.f;
  float hi = (k + 1 < Kreal) ? W[n * Kreal + k + 1] : 0.f;
  wsU[idx] = packh2(lo, hi);
}

// ---------------- WMMA fragment helpers (wave32, 16x16x32 f16) ----------------
__device__ __forceinline__ v8f wmma_f16(v16h a, v16h b, v8f c) {
  return __builtin_amdgcn_wmma_f32_16x16x32_f16(false, a, false, b, (short)0, c, false, false);
}

// A fragment 16x32 from row-major f16 [M][strideH]: two contiguous 16B blocks per lane.
// dwords 0..7 = k in [kbase+16*grp, +16)
__device__ __forceinline__ v16h ldsA(const _Float16* base, int row0, int strideH, int kbase, int lane) {
  union { v16h v; v4u q[2]; } u;
  const _Float16* rp = base + (row0 + (lane & 15)) * strideH + kbase + 16 * (lane >> 4);
  u.q[0] = *(const v4u*)rp;
  u.q[1] = *(const v4u*)(rp + 8);
  return u.v;
}

// B fragment 32x16 where B(k,n) = M[n][k] (row-major [N][strideH]):
// lane l covers k in [kbase + 16*grp, +16) -> one contiguous 32B run.
__device__ __forceinline__ v16h ldsBT(const _Float16* base, int n0, int strideH, int kbase, int lane) {
  union { v16h v; v4u q[2]; } u;
  const _Float16* rp = base + (n0 + (lane & 15)) * strideH + kbase + 16 * (lane >> 4);
  u.q[0] = *(const v4u*)rp;
  u.q[1] = *(const v4u*)(rp + 8);
  return u.v;
}

// B fragment from fragment-native global weight panel (32B contiguous per lane).
__device__ __forceinline__ v16h gBfrag(const uint32_t* __restrict__ wp, int f, int lane) {
  union { v16h v; v4u q[2]; } u;
  const v4u* p = (const v4u*)(wp + f * 256 + lane * 8);
  u.q[0] = p[0];
  u.q[1] = p[1];
  return u.v;
}

// C fragment (f32) + bias -> f16 row-major LDS store (8 b16 stores, fixed col per lane)
__device__ __forceinline__ void storeC_rowmajor(_Float16* dst, int row0, int strideH,
                                                int n0, v8f c, float bb, int grp, int ln16) {
#pragma unroll
  for (int i = 0; i < 8; ++i)
    dst[(row0 + i + 8 * grp) * strideH + n0 + ln16] = (_Float16)(c[i] + bb);
}

// ---------------- fused forward: one wave handles 16 batch elements ----------------
__global__ __launch_bounds__(32)
void fused_net(const float* __restrict__ states, const float* __restrict__ action,
               const int* __restrict__ block_id, const uint32_t* __restrict__ wsU,
               const float* __restrict__ bq, const float* __restrict__ bk,
               const float* __restrict__ bv, const float* __restrict__ b1,
               const float* __restrict__ b2, const float* __restrict__ Wc1,
               const float* __restrict__ bc1, const float* __restrict__ Wc2,
               const float* __restrict__ bc2, float* __restrict__ out) {
  __shared__ alignas(16) char smem[59776];
  _Float16* baH  = (_Float16*)(smem + 0);       // [48][32]   ba, K padded
  _Float16* qH   = (_Float16*)(smem + 3072);    // [48][128]  q      ; later: res
  _Float16* kH   = (_Float16*)(smem + 15360);   // [48][128]  k      ; later: tanh(resW1)
  _Float16* vT   = (_Float16*)(smem + 27648);   // [128][64]  v transposed (vT[n][k]), k 48..63 zero
  _Float16* aH   = (_Float16*)(smem + 44032);   // [48][64]   attn A ; later: flat f32 [16][75]
  float*    numF = (float*)   (smem + 50176);   // [48][48]
  float*    qn   = (float*)   (smem + 59392);   // [48]
  float*    kn   = (float*)   (smem + 59584);   // [48]

  const int lane = threadIdx.x;
  const int grp  = lane >> 4;
  const int ln16 = lane & 15;
  const int b0   = blockIdx.x * BT;

  if (lane == 0) {
    __builtin_prefetch(wsU + WS_WQ, 0, 1);
    __builtin_prefetch(wsU + WS_W1, 0, 1);
    __builtin_prefetch(wsU + WS_W2, 0, 1);
  }

  // ---- Stage 1: build ba (f16); zero vT pad columns k=48..63 ----
  for (int idx = lane; idx < ROWS * 32; idx += 32) {
    int r = idx >> 5, c = idx & 31;
    int gb = b0 + r / 3, blk = r % 3;
    float val = 0.f;
    if (c < SSZ)      val = states[gb * (NBLK * SSZ) + blk * SSZ + c];
    else if (c < SA)  val = (block_id[gb * NBLK + blk] == 1) ? action[gb * NACT + (c - SSZ)] : -1.0f;
    baH[idx] = (_Float16)val;
  }
  for (int idx = lane; idx < VD * 8; idx += 32) {      // 128 rows x 8 dwords (16 halves)
    int n = idx >> 3, dw = idx & 7;
    ((uint32_t*)(vT + n * 64 + 48))[dw] = 0u;
  }
  __syncthreads();

  // ---- Stage 2a: q,k = ba @ W^T + b (WMMA, K=32); shared A fragment per mt ----
#pragma unroll
  for (int mt = 0; mt < 3; ++mt) {
    v16h a = ldsA(baH, 16 * mt, 32, 0, lane);
    for (int nt = 0; nt < 8; ++nt) {
      v16h wq = gBfrag(wsU + WS_WQ, nt, lane);
      v8f cq = {};
      cq = wmma_f16(a, wq, cq);
      storeC_rowmajor(qH, 16 * mt, VD, 16 * nt, cq, bq[16 * nt + ln16], grp, ln16);
      v16h wk = gBfrag(wsU + WS_WK, nt, lane);
      v8f ck = {};
      ck = wmma_f16(a, wk, ck);
      storeC_rowmajor(kH, 16 * mt, VD, 16 * nt, ck, bk[16 * nt + ln16], grp, ln16);
    }
  }

  // ---- Stage 2b: v = ba @ Wv^T + bv, staged transposed (pair-packed dword stores) ----
#pragma unroll
  for (int mt = 0; mt < 3; ++mt) {
    v16h a = ldsA(baH, 16 * mt, 32, 0, lane);
    for (int nt = 0; nt < 8; ++nt) {
      v16h wv = gBfrag(wsU + WS_WV, nt, lane);
      v8f c = {};
      c = wmma_f16(a, wv, c);
      float bb = bv[16 * nt + ln16];
      uint32_t* dp = (uint32_t*)(vT + (16 * nt + ln16) * 64 + 16 * mt + 8 * grp);
#pragma unroll
      for (int i = 0; i < 4; ++i)
        dp[i] = packh2(c[2 * i] + bb, c[2 * i + 1] + bb);
    }
  }
  __syncthreads();

  // ---- Stage 3: row norms of q and k ----
  for (int r = lane; r < ROWS; r += 32) {
    const uint32_t* qr = (const uint32_t*)(qH + r * VD);
    const uint32_t* kr = (const uint32_t*)(kH + r * VD);
    float sq = 0.f, sk = 0.f;
    for (int dw = 0; dw < VD / 2; ++dw) {
      H2 a; a.u = qr[dw];
      H2 b; b.u = kr[dw];
      float a0 = (float)a.h[0], a1 = (float)a.h[1];
      float c0 = (float)b.h[0], c1 = (float)b.h[1];
      sq += a0 * a0 + a1 * a1;
      sk += c0 * c0 + c1 * c1;
    }
    qn[r] = sqrtf(sq); kn[r] = sqrtf(sk);
  }

  // ---- Stage 4: num = Q @ K^T (WMMA, K=128) ----
  for (int mt = 0; mt < 3; ++mt)
    for (int jt = 0; jt < 3; ++jt) {
      v8f c = {};
#pragma unroll
      for (int ks = 0; ks < 4; ++ks) {
        v16h a = ldsA(qH, 16 * mt, VD, 32 * ks, lane);
        v16h b = ldsBT(kH, 16 * jt, VD, 32 * ks, lane);
        c = wmma_f16(a, b, c);
      }
#pragma unroll
      for (int i = 0; i < 8; ++i)
        numF[(16 * mt + i + 8 * grp) * ROWS + 16 * jt + ln16] = c[i];
    }
  __syncthreads();

  // ---- Stage 5: cosine + softmax -> block-diagonal attention A [48][64] ----
  {
    uint32_t* az = (uint32_t*)aH;
    for (int idx = lane; idx < ROWS * 64 / 2; idx += 32) az[idx] = 0u;
  }
  __syncthreads();
  for (int r = lane; r < ROWS; r += 32) {
    int c0 = 3 * (r / 3);
    float d0 = numF[r * ROWS + c0 + 0] / fmaxf(qn[r] * kn[c0 + 0], 1e-8f);
    float d1 = numF[r * ROWS + c0 + 1] / fmaxf(qn[r] * kn[c0 + 1], 1e-8f);
    float d2 = numF[r * ROWS + c0 + 2] / fmaxf(qn[r] * kn[c0 + 2], 1e-8f);
    float m  = fmaxf(d0, fmaxf(d1, d2));
    float e0 = expf(d0 - m), e1 = expf(d1 - m), e2 = expf(d2 - m);
    float inv = 1.f / (e0 + e1 + e2);
    aH[r * 64 + c0 + 0] = (_Float16)(e0 * inv);
    aH[r * 64 + c0 + 1] = (_Float16)(e1 * inv);
    aH[r * 64 + c0 + 2] = (_Float16)(e2 * inv);
  }
  __syncthreads();

  // ---- Stage 6: res = A @ V (WMMA, K=64, B from vT) -> qH ----
  for (int mt = 0; mt < 3; ++mt)
    for (int nt = 0; nt < 8; ++nt) {
      v8f c = {};
#pragma unroll
      for (int ks = 0; ks < 2; ++ks) {
        v16h a = ldsA(aH, 16 * mt, 64, 32 * ks, lane);
        v16h b = ldsBT(vT, 16 * nt, 64, 32 * ks, lane);
        c = wmma_f16(a, b, c);
      }
      storeC_rowmajor(qH, 16 * mt, VD, 16 * nt, c, 0.f, grp, ln16);
    }
  __syncthreads();

  // ---- Stage 7: tanh(res @ W1^T + b1) -> kH (WMMA, K=128) ----
  for (int mt = 0; mt < 3; ++mt)
    for (int nt = 0; nt < 8; ++nt) {
      v8f c = {};
#pragma unroll
      for (int ks = 0; ks < 4; ++ks) {
        v16h a = ldsA(qH, 16 * mt, VD, 32 * ks, lane);
        v16h b = gBfrag(wsU + WS_W1, nt * 4 + ks, lane);
        c = wmma_f16(a, b, c);
      }
      float bb = b1[16 * nt + ln16];
#pragma unroll
      for (int i = 0; i < 8; ++i)
        kH[(16 * mt + i + 8 * grp) * VD + 16 * nt + ln16] = (_Float16)tanhf(c[i] + bb);
    }
  __syncthreads();

  // ---- Stage 8: out[:, :48] = states + tanhres @ W2^T + b2 (WMMA, K=128) ----
  for (int mt = 0; mt < 3; ++mt) {
    v8f c = {};
#pragma unroll
    for (int ks = 0; ks < 4; ++ks) {
      v16h a = ldsA(kH, 16 * mt, VD, 32 * ks, lane);
      v16h b = gBfrag(wsU + WS_W2, ks, lane);
      c = wmma_f16(a, b, c);
    }
#pragma unroll
    for (int i = 0; i < 8; ++i) {
      int row = 16 * mt + i + 8 * grp;
      int bl  = row / 3, blk = row % 3;
      int gb  = b0 + bl;
      float val = c[i] + b2[ln16] + states[gb * (NBLK * SSZ) + blk * SSZ + ln16];
      out[gb * OUTD + blk * SSZ + ln16] = val;
    }
  }

  // ---- Stage 9: confidence head (f32 VALU; flat staged in freed aH space) ----
  __syncthreads();
  float* flatF = (float*)aH;               // [16][75]
  for (int idx = lane; idx < BT * 75; idx += 32) {
    int bl = idx / 75, t = idx % 75;
    int blk = t / SA, c = t % SA;
    int gb = b0 + bl;
    float val;
    if (c < SSZ) val = states[gb * (NBLK * SSZ) + blk * SSZ + c];
    else         val = (block_id[gb * NBLK + blk] == 1) ? action[gb * NACT + (c - SSZ)] : -1.0f;
    flatF[idx] = val;
  }
  __syncthreads();
  if (lane < BT) {
    int gb = b0 + lane;
    const float* fl = flatF + lane * 75;
    float h[10];
#pragma unroll
    for (int c = 0; c < 10; ++c) {
      float s = bc1[c];
      for (int t = 0; t < 75; ++t) s += fl[t] * Wc1[c * 75 + t];
      h[c] = 1.f / (1.f + expf(-s));
    }
#pragma unroll
    for (int o = 0; o < 3; ++o) {
      float s = bc2[o];
#pragma unroll
      for (int c = 0; c < 10; ++c) s += h[c] * Wc2[o * 10 + c];
      out[gb * OUTD + 48 + o] = 1.f / (1.f + expf(-s));
    }
  }
}

extern "C" void kernel_launch(void* const* d_in, const int* in_sizes, int n_in,
                              void* d_out, int out_size, void* d_ws, size_t ws_size,
                              hipStream_t stream) {
  (void)in_sizes; (void)n_in; (void)out_size; (void)ws_size;
  const float* states   = (const float*)d_in[0];
  const float* action   = (const float*)d_in[1];
  const int*   block_id = (const int*)  d_in[2];
  const float* Wq  = (const float*)d_in[3],  *bq  = (const float*)d_in[4];
  const float* Wk  = (const float*)d_in[5],  *bk  = (const float*)d_in[6];
  const float* Wv  = (const float*)d_in[7],  *bv  = (const float*)d_in[8];
  const float* W1  = (const float*)d_in[9],  *b1  = (const float*)d_in[10];
  const float* W2  = (const float*)d_in[11], *b2  = (const float*)d_in[12];
  const float* Wc1 = (const float*)d_in[13], *bc1 = (const float*)d_in[14];
  const float* Wc2 = (const float*)d_in[15], *bc2 = (const float*)d_in[16];
  float*    out = (float*)d_out;
  uint32_t* wsU = (uint32_t*)d_ws;

  prep_weights<<<(WS_TOTAL + 255) / 256, 256, 0, stream>>>(Wq, Wk, Wv, W1, W2, wsU);
  fused_net<<<B_TOTAL / BT, 32, 0, stream>>>(states, action, block_id, wsU,
                                             bq, bk, bv, b1, b2, Wc1, bc1, Wc2, bc2, out);
}
